// MambaLM_13434657702442
// MI455X (gfx1250) — matrix operand order repaired and checked by reading, not compiled, not run on hardware
//
#include <hip/hip_runtime.h>
#include <hip/hip_bf16.h>

typedef __attribute__((ext_vector_type(16))) _Float16 v16h;
typedef __attribute__((ext_vector_type(8)))  _Float16 v8h;
typedef __attribute__((ext_vector_type(4)))  _Float16 v4h;
typedef __attribute__((ext_vector_type(8)))  float    v8f;
typedef __attribute__((ext_vector_type(4)))  unsigned int u32x4;
typedef __attribute__((ext_vector_type(4)))  int          i32x4;
typedef __attribute__((ext_vector_type(8)))  int          i32x8;

#define B_  16
#define L_  2048
#define D_  512
#define V_  512
#define NL_ 4
#define M_  (B_*L_)

// LDS row stride for the staged weight strip: 512 + 8 halfs (1040 B = 260
// words -> 4 banks/row), so a B-tile's 16 lanes hit disjoint 4-bank groups.
#define SWP 520
// LDS row stride (floats) for the f32 output bounce buffer.
#define ORS 68

static __device__ __forceinline__ float sigmoidf_(float x) {
    return 1.0f / (1.0f + __expf(-x));
}

static __device__ __forceinline__ v16h cat8(v8h lo, v8h hi) {
    return __builtin_shufflevector(lo, hi, 0,1,2,3,4,5,6,7,8,9,10,11,12,13,14,15);
}

#define WMMA_F16(A, Bm, Cacc) \
    __builtin_amdgcn_wmma_f32_16x16x32_f16(false, (A), false, (Bm), (short)0, (Cacc), false, false)

// ---------------------------------------------------------------------------
// TDM: stage a 64x512 f16 strip (global, row stride 512) into LDS with
// +8-half padding per row (pad_interval = 256 DWORDs, pad_amount = 4 DWORDs),
// i.e. the SWP=520 layout the ds_load_b128 B-reads expect.
// D# per CDNA5 ISA: group0 = {flags, lds_addr, global_addr, type=2},
// group1 = {mode bits, dims/tiles/strides}. 2D tensor -> groups 2/3 zero.
// ---------------------------------------------------------------------------
static __device__ __forceinline__ void tdm_stage_strip(const _Float16* gsrc,
                                                       _Float16* lds_dst) {
    unsigned long long ga = (unsigned long long)(size_t)(const void*)gsrc;
    unsigned int ldsoff = (unsigned int)(size_t)(void*)lds_dst; // flat[31:0] == LDS offset
    u32x4 g0;
    g0[0] = 1u;                                   // count=1, user descriptor
    g0[1] = ldsoff;                               // lds_addr
    g0[2] = (unsigned int)(ga & 0xFFFFFFFFu);     // global_addr[31:0]
    g0[3] = (unsigned int)((ga >> 32) & 0x1FFFFFFu) | (2u << 30); // [56:32] | type=2
    i32x8 g1;
    g1[0] = (1 << 16)    // data_size = 2 bytes
          | (1 << 20)    // pad_enable (LDS row padding on load)
          | (7 << 22)    // pad_interval: 256 DWORDs = one 512-half row
          | (3 << 25);   // pad_amount:  4 DWORDs   = 8 halfs
    g1[1] = (int)(512u << 16);   // tensor_dim0 = 512 (halfs per row)
    g1[2] = (int)(64u  << 16);   // tensor_dim1 = 64 rows
    g1[3] = (int)(512u << 16);   // tile_dim0 = 512
    g1[4] = 64;                  // tile_dim1 = 64, tile_dim2 = 0
    g1[5] = 512;                 // tensor_dim0_stride = 512
    g1[6] = 0;
    g1[7] = 0;
    i32x4 z4 = {0, 0, 0, 0};
#if __clang_major__ >= 23
    i32x8 z8 = {0, 0, 0, 0, 0, 0, 0, 0};
    __builtin_amdgcn_tensor_load_to_lds(g0, g1, z4, z4, z8, 0);
#else
    __builtin_amdgcn_tensor_load_to_lds(g0, g1, z4, z4, 0);
#endif
}

// ---------------------------------------------------------------------------
// one-hot embedding: y[m, d] = (d == x[m]) ? 1 : 0   (residual stream, f32)
// ---------------------------------------------------------------------------
__global__ void onehot_kernel(const int* __restrict__ x, float* __restrict__ y) {
    int m = blockIdx.x;
    int tok = x[m];
    float* row = y + (size_t)m * D_;
    for (int d = threadIdx.x; d < D_; d += blockDim.x)
        row[d] = (d == tok) ? 1.0f : 0.0f;
}

// ---------------------------------------------------------------------------
// LayerNorm, one wave per row; f32 in, f16 out (feeds a WMMA GEMM)
// ---------------------------------------------------------------------------
__global__ void layernorm_kernel(const float* __restrict__ X,
                                 const float* __restrict__ g,
                                 const float* __restrict__ b,
                                 _Float16* __restrict__ Y, int M) {
    int wid  = (blockIdx.x * blockDim.x + threadIdx.x) >> 5;
    int lane = threadIdx.x & 31;
    if (wid >= M) return;
    const float* row = X + (size_t)wid * D_;
    float s = 0.f, ss = 0.f;
#pragma unroll
    for (int j = 0; j < D_/32; ++j) {
        float v = row[lane + 32*j];
        s += v; ss += v*v;
    }
#pragma unroll
    for (int o = 16; o > 0; o >>= 1) {
        s  += __shfl_xor(s,  o, 32);
        ss += __shfl_xor(ss, o, 32);
    }
    float mean = s * (1.0f/D_);
    float var  = ss * (1.0f/D_) - mean*mean;
    float inv  = rsqrtf(var + 1e-5f);
    _Float16* out = Y + (size_t)wid * D_;
#pragma unroll
    for (int j = 0; j < D_/32; ++j) {
        int d = lane + 32*j;
        out[d] = (_Float16)((row[d] - mean) * inv * g[d] + b[d]);
    }
}

// ---------------------------------------------------------------------------
// Weight repacks f32 -> f16
// ---------------------------------------------------------------------------
__global__ void repack_f16_kernel(const float* __restrict__ src,
                                  _Float16* __restrict__ dst, int n) {
    int i = blockIdx.x * blockDim.x + threadIdx.x;
    if (i < n) dst[i] = (_Float16)src[i];
}

// conv_w [O][I][3] f32 -> dst [3][O][I] f16 (three contiguous [N,K] mats)
__global__ void repack_conv_kernel(const float* __restrict__ src,
                                   _Float16* __restrict__ dst) {
    int i = blockIdx.x * blockDim.x + threadIdx.x;
    if (i >= 3*D_*D_) return;
    int k   = i / (D_*D_);
    int rem = i - k*(D_*D_);
    int o   = rem / D_;
    int c   = rem - o*D_;
    dst[i] = (_Float16)src[(size_t)(o*D_ + c)*3 + k];
}

// ---------------------------------------------------------------------------
// Accumulator tile -> LDS bounce buffer (bias applied here; lane-resident n)
// rows within block: wrow0 = wave*16 + half*8;  column within block: ncol
// ---------------------------------------------------------------------------
__device__ __forceinline__ void acc_to_lds(float* __restrict__ swf,
                                           const float* __restrict__ bias,
                                           v8f acc, int wrow0, int ncol, int nglob) {
    float bn = bias ? bias[nglob] : 0.0f;
#pragma unroll
    for (int r = 0; r < 8; ++r)
        swf[(wrow0 + r) * ORS + ncol] = acc[r] + bn;
}

// LDS bounce buffer -> global, coalesced float4 / 4xf16 rows (+resid here)
__device__ __forceinline__ void lds_to_global(const float* __restrict__ swf,
                                              const float* __restrict__ resid,
                                              float* __restrict__ Cf,
                                              _Float16* __restrict__ Ch,
                                              int mrow0, int n0, int N, int tid) {
    for (int i = tid; i < 128*16; i += 256) {
        int row = i >> 4, c = (i & 15) << 2;
        const float* src = swf + row * ORS + c;
        size_t gidx = (size_t)(mrow0 + row) * N + n0 + c;
        float v0 = src[0], v1 = src[1], v2 = src[2], v3 = src[3];
        if (resid) {
            v0 += resid[gidx+0]; v1 += resid[gidx+1];
            v2 += resid[gidx+2]; v3 += resid[gidx+3];
        }
        if (Cf) {
            float4 o; o.x = v0; o.y = v1; o.z = v2; o.w = v3;
            *(float4*)&Cf[gidx] = o;
        } else {
            v4h o = {(_Float16)v0, (_Float16)v1, (_Float16)v2, (_Float16)v3};
            *(v4h*)&Ch[gidx] = o;
        }
    }
}

// ---------------------------------------------------------------------------
// GEMM: C[M,N] = A[M,512](f16) x W[N,512](f16)^T + bias (+resid)
// Block = 8 waves = 128 rows x 64 cols.  W strip staged via TDM into LDS;
// A loads software-pipelined from global; B via ds_load_b128; C bounced
// through LDS for coalesced stores.
// ---------------------------------------------------------------------------
__global__ __launch_bounds__(256)
void gemm_kernel(const _Float16* __restrict__ A,
                 const _Float16* __restrict__ W,
                 const float* __restrict__ bias,
                 const float* __restrict__ resid,
                 float* __restrict__ Cf, _Float16* __restrict__ Ch,
                 int M, int N) {
    __shared__ _Float16 sw[64 * SWP];
    int stripsN = N >> 6;
    int blkM = blockIdx.x / stripsN;
    int blkN = blockIdx.x - blkM * stripsN;
    int n0 = blkN << 6;

    // Wave 0 issues the tensor DMA (EXEC-independent, one issue per wave),
    // drains TENSORcnt, then the barrier publishes LDS to all 8 waves.
    if (threadIdx.x < 32) {
        tdm_stage_strip(W + (size_t)n0 * D_, sw);
        __builtin_amdgcn_s_wait_tensorcnt(0);
    }
    __syncthreads();

    int wave = threadIdx.x >> 5, lane = threadIdx.x & 31;
    int half = lane >> 4, l16 = lane & 15;
    int m0 = (blkM << 7) + (wave << 4);
    const _Float16* arow = A + (size_t)(m0 + l16) * D_;
    __builtin_prefetch(arow + 256, 0, 1);   // global_prefetch_b8, 2nd half of row

    v8f acc0 = {}, acc1 = {}, acc2 = {}, acc3 = {};
    v8h alo = *(const v8h*)(arow + half*16);
    v8h ahi = *(const v8h*)(arow + 8 + half*16);
    for (int k0 = 0; k0 < D_; k0 += 32) {
        v8h nlo = {}, nhi = {};
        if (k0 + 32 < D_) {            // pipeline next A chunk over the WMMAs
            nlo = *(const v8h*)(arow + k0 + 32 + half*16);
            nhi = *(const v8h*)(arow + k0 + 40 + half*16);
        }
        v16h a = cat8(alo, ahi);
        const _Float16* sb = &sw[(size_t)l16 * SWP + k0 + half*16];
        v16h b0 = cat8(*(const v8h*)(sb),           *(const v8h*)(sb + 8));
        v16h b1 = cat8(*(const v8h*)(sb + 16*SWP),  *(const v8h*)(sb + 16*SWP + 8));
        v16h b2 = cat8(*(const v8h*)(sb + 32*SWP),  *(const v8h*)(sb + 32*SWP + 8));
        v16h b3 = cat8(*(const v8h*)(sb + 48*SWP),  *(const v8h*)(sb + 48*SWP + 8));
        acc0 = WMMA_F16(a, b0, acc0);
        acc1 = WMMA_F16(a, b1, acc1);
        acc2 = WMMA_F16(a, b2, acc2);
        acc3 = WMMA_F16(a, b3, acc3);
        alo = nlo; ahi = nhi;
    }

    __syncthreads();                    // all waves done reading weights
    float* swf = (float*)sw;            // reuse LDS as f32 bounce buffer
    int wrow0 = (wave << 4) + half*8;
    acc_to_lds(swf, bias, acc0, wrow0,  0 + l16, n0 +  0 + l16);
    acc_to_lds(swf, bias, acc1, wrow0, 16 + l16, n0 + 16 + l16);
    acc_to_lds(swf, bias, acc2, wrow0, 32 + l16, n0 + 32 + l16);
    acc_to_lds(swf, bias, acc3, wrow0, 48 + l16, n0 + 48 + l16);
    __syncthreads();
    lds_to_global(swf, resid, Cf, Ch, blkM << 7, n0, N, threadIdx.x);
}

// ---------------------------------------------------------------------------
// Conv1d(k=3, pad=1) as 3 shifted GEMMs; W3 = f16 [3][N][512].
// Each tap's strip staged via TDM; f16 A, f32 C (bounced through LDS).
// ---------------------------------------------------------------------------
__global__ __launch_bounds__(256)
void conv_gemm_kernel(const _Float16* __restrict__ A,
                      const _Float16* __restrict__ W3,
                      const float* __restrict__ bias,
                      float* __restrict__ C, int M, int N) {
    __shared__ _Float16 sw[64 * SWP];
    int stripsN = N >> 6;
    int blkM = blockIdx.x / stripsN;
    int blkN = blockIdx.x - blkM * stripsN;
    int n0 = blkN << 6;
    int wave = threadIdx.x >> 5, lane = threadIdx.x & 31;
    int half = lane >> 4, l16 = lane & 15;
    int m0 = (blkM << 7) + (wave << 4);
    int mrow = m0 + l16;
    int l = mrow & (L_ - 1);

    v8f acc0 = {}, acc1 = {}, acc2 = {}, acc3 = {};
    for (int s = 0; s < 3; ++s) {
        __syncthreads();   // all waves done reading previous tap's strip
        if (threadIdx.x < 32) {
            tdm_stage_strip(W3 + ((size_t)s * N + n0) * D_, sw);
            __builtin_amdgcn_s_wait_tensorcnt(0);
        }
        __syncthreads();

        bool valid = (unsigned)(l + s - 1) < (unsigned)L_;
        const _Float16* arow = A + (size_t)(mrow + s - 1) * D_;
        v8h alo = {}, ahi = {};
        if (valid) {
            alo = *(const v8h*)(arow + half*16);
            ahi = *(const v8h*)(arow + 8 + half*16);
        }
        for (int k0 = 0; k0 < D_; k0 += 32) {
            v8h nlo = {}, nhi = {};
            if (valid && (k0 + 32 < D_)) {
                nlo = *(const v8h*)(arow + k0 + 32 + half*16);
                nhi = *(const v8h*)(arow + k0 + 40 + half*16);
            }
            v16h a = cat8(alo, ahi);
            const _Float16* sb = &sw[(size_t)l16 * SWP + k0 + half*16];
            v16h b0 = cat8(*(const v8h*)(sb),           *(const v8h*)(sb + 8));
            v16h b1 = cat8(*(const v8h*)(sb + 16*SWP),  *(const v8h*)(sb + 16*SWP + 8));
            v16h b2 = cat8(*(const v8h*)(sb + 32*SWP),  *(const v8h*)(sb + 32*SWP + 8));
            v16h b3 = cat8(*(const v8h*)(sb + 48*SWP),  *(const v8h*)(sb + 48*SWP + 8));
            acc0 = WMMA_F16(a, b0, acc0);
            acc1 = WMMA_F16(a, b1, acc1);
            acc2 = WMMA_F16(a, b2, acc2);
            acc3 = WMMA_F16(a, b3, acc3);
            alo = nlo; ahi = nhi;
        }
    }

    __syncthreads();
    float* swf = (float*)sw;
    int wrow0 = (wave << 4) + half*8;
    acc_to_lds(swf, bias, acc0, wrow0,  0 + l16, n0 +  0 + l16);
    acc_to_lds(swf, bias, acc1, wrow0, 16 + l16, n0 + 16 + l16);
    acc_to_lds(swf, bias, acc2, wrow0, 32 + l16, n0 + 32 + l16);
    acc_to_lds(swf, bias, acc3, wrow0, 48 + l16, n0 + 48 + l16);
    __syncthreads();
    lds_to_global(swf, nullptr, C, nullptr, blkM << 7, n0, N, threadIdx.x);
}

// ---------------------------------------------------------------------------
// SSM gate: g[m] = sigmoid(dot(Z[m,:], wg) + bg).  One wave per row (f32 in).
// ---------------------------------------------------------------------------
__global__ void gate_kernel(const float* __restrict__ Z,
                            const float* __restrict__ wg,
                            const float* __restrict__ bg,
                            float* __restrict__ g, int M) {
    int wid  = (blockIdx.x * blockDim.x + threadIdx.x) >> 5;
    int lane = threadIdx.x & 31;
    if (wid >= M) return;
    const float* row = Z + (size_t)wid * D_;
    float s = 0.f;
#pragma unroll
    for (int j = 0; j < D_/32; ++j)
        s += row[lane + 32*j] * wg[lane + 32*j];
#pragma unroll
    for (int o = 16; o > 0; o >>= 1) s += __shfl_xor(s, o, 32);
    if (lane == 0) g[wid] = sigmoidf_(s + bg[0]);
}

// ---------------------------------------------------------------------------
// Selective scan: h = (1-g)h + g*z over L (f32 state), f16 output (GEMM A).
// ---------------------------------------------------------------------------
__global__ void scan_kernel(const float* __restrict__ Z,
                            const float* __restrict__ g,
                            _Float16* __restrict__ H) {
    int t = blockIdx.x * blockDim.x + threadIdx.x;
    if (t >= B_*D_) return;
    int b = t / D_, d = t - b*D_;
    const float* zrow = Z + (size_t)b * L_ * D_ + d;
    _Float16*    hrow = H + (size_t)b * L_ * D_ + d;
    const float* grow = g + (size_t)b * L_;
    float h = 0.0f;
    for (int l = 0; l < L_; ++l) {
        float gv = grow[l];
        h = (1.0f - gv) * h + gv * zrow[(size_t)l * D_];
        hrow[(size_t)l * D_] = (_Float16)h;
    }
}

// ---------------------------------------------------------------------------
// GLU: out[m,d] = sigmoid(fc[m,d]) * fc[m, D_+d]; f16 out (GEMM A)
// ---------------------------------------------------------------------------
__global__ void glu_kernel(const float* __restrict__ FC,
                           _Float16* __restrict__ Y, int M) {
    int i = blockIdx.x * blockDim.x + threadIdx.x;
    if (i >= M * D_) return;
    int m = i / D_, d = i - m*D_;
    const float* row = FC + (size_t)m * (2*D_);
    Y[i] = (_Float16)(sigmoidf_(row[d]) * row[D_ + d]);
}

// ---------------------------------------------------------------------------
// Host launcher
// ---------------------------------------------------------------------------
extern "C" void kernel_launch(void* const* d_in, const int* in_sizes, int n_in,
                              void* d_out, int out_size, void* d_ws, size_t ws_size,
                              hipStream_t stream) {
    const int*   x      = (const int*)  d_in[0];
    const float* ln_g   = (const float*)d_in[1];
    const float* ln_b   = (const float*)d_in[2];
    const float* conv_w = (const float*)d_in[3];
    const float* conv_b = (const float*)d_in[4];
    const float* wg     = (const float*)d_in[5];
    const float* bg     = (const float*)d_in[6];
    const float* wout   = (const float*)d_in[7];
    const float* bout   = (const float*)d_in[8];
    const float* fc_w   = (const float*)d_in[9];
    const float* fc_b   = (const float*)d_in[10];
    const float* mo_w   = (const float*)d_in[11];
    const float* mo_b   = (const float*)d_in[12];
    const float* lnf_g  = (const float*)d_in[13];
    const float* lnf_b  = (const float*)d_in[14];
    const float* head_w = (const float*)d_in[15];
    const float* head_b = (const float*)d_in[16];
    float* out = (float*)d_out;

    char* ws = (char*)d_ws;
    size_t off = 0;
    auto wsalloc = [&](size_t bytes) -> void* {
        void* p = ws + off;
        off += (bytes + 255) & ~(size_t)255;
        return p;
    };
    float*     y    = (float*)    wsalloc((size_t)M_ * D_   * sizeof(float));     // residual
    float*     t2   = (float*)    wsalloc((size_t)M_ * D_   * sizeof(float));     // conv out (f32)
    float*     fcb  = (float*)    wsalloc((size_t)M_ * 2*D_ * sizeof(float));     // fc out (f32)
    float*     gbuf = (float*)    wsalloc((size_t)M_        * sizeof(float));
    _Float16*  t1h  = (_Float16*) wsalloc((size_t)M_ * D_   * sizeof(_Float16));  // f16 act A
    _Float16*  t2h  = (_Float16*) wsalloc((size_t)M_ * D_   * sizeof(_Float16));  // f16 act B
    _Float16*  wbuf = (_Float16*) wsalloc((size_t)3*D_*D_   * sizeof(_Float16));  // f16 weights

    const int TPB = 256;
    auto gemm_blocks = [](int M, int N) { return (M >> 7) * (N >> 6); };

    onehot_kernel<<<M_, TPB, 0, stream>>>(x, y);

    for (int i = 0; i < NL_; ++i) {
        // z = LN(y) -> f16
        layernorm_kernel<<<(M_*32 + TPB-1)/TPB, TPB, 0, stream>>>(
            y, ln_g + i*D_, ln_b + i*D_, t1h, M_);
        // conv(k=3,pad=1) channel-mixing as 3 shifted GEMMs -> f32
        repack_conv_kernel<<<(3*D_*D_ + TPB-1)/TPB, TPB, 0, stream>>>(
            conv_w + (size_t)i*D_*D_*3, wbuf);
        conv_gemm_kernel<<<gemm_blocks(M_, D_), TPB, 0, stream>>>(
            t1h, wbuf, conv_b + i*D_, t2, M_, D_);
        // scalar selective gate + scan (f32 state, f16 out)
        gate_kernel<<<(M_*32 + TPB-1)/TPB, TPB, 0, stream>>>(
            t2, wg + i*D_, bg + i, gbuf, M_);
        scan_kernel<<<(B_*D_ + TPB-1)/TPB, TPB, 0, stream>>>(t2, gbuf, t1h);
        // out-projection of scan states -> f16 (consumed by fc GEMM)
        repack_f16_kernel<<<(D_*D_ + TPB-1)/TPB, TPB, 0, stream>>>(
            wout + (size_t)i*D_*D_, wbuf, D_*D_);
        gemm_kernel<<<gemm_blocks(M_, D_), TPB, 0, stream>>>(
            t1h, wbuf, bout + i*D_, nullptr, nullptr, t2h, M_, D_);
        // FC D -> 2D -> f32
        repack_f16_kernel<<<(2*D_*D_ + TPB-1)/TPB, TPB, 0, stream>>>(
            fc_w + (size_t)i*2*D_*D_, wbuf, 2*D_*D_);
        gemm_kernel<<<gemm_blocks(M_, 2*D_), TPB, 0, stream>>>(
            t2h, wbuf, fc_b + i*2*D_, nullptr, fcb, nullptr, M_, 2*D_);
        // GLU -> f16
        glu_kernel<<<((M_*D_) + TPB-1)/TPB, TPB, 0, stream>>>(fcb, t1h, M_);
        // output projection + residual into y (f32)
        repack_f16_kernel<<<(D_*D_ + TPB-1)/TPB, TPB, 0, stream>>>(
            mo_w + (size_t)i*D_*D_, wbuf, D_*D_);
        gemm_kernel<<<gemm_blocks(M_, D_), TPB, 0, stream>>>(
            t1h, wbuf, mo_b + i*D_, /*resid=*/y, y, nullptr, M_, D_);
    }

    // final LN + head
    layernorm_kernel<<<(M_*32 + TPB-1)/TPB, TPB, 0, stream>>>(
        y, lnf_g, lnf_b, t1h, M_);
    repack_f16_kernel<<<(V_*D_ + TPB-1)/TPB, TPB, 0, stream>>>(
        head_w, wbuf, V_*D_);
    gemm_kernel<<<gemm_blocks(M_, V_), TPB, 0, stream>>>(
        t1h, wbuf, head_b, nullptr, out, nullptr, M_, V_);
}
